// ViT_54331336294408
// MI455X (gfx1250) — matrix operand order repaired and checked by reading, not compiled, not run on hardware
//
#include <hip/hip_runtime.h>
#include <math.h>

// ---------------------------------------------------------------------------
// Types
// ---------------------------------------------------------------------------
typedef __attribute__((ext_vector_type(16))) __bf16          v16bf;
typedef __attribute__((ext_vector_type(8)))  float           v8f;
typedef __attribute__((ext_vector_type(8)))  unsigned short  u16x8;
typedef __attribute__((ext_vector_type(16))) unsigned short  u16x16;

#define VIT_B   32
#define VIT_S   197
#define VIT_E   768
#define VIT_H   12
#define VIT_HD  64
#define VIT_M   3072
#define VIT_L   12
#define VIT_N   196
#define ROWS    (VIT_B * VIT_S)        // 6304
#define PROWS   (VIT_B * VIT_N)        // 6272

// ---------------------------------------------------------------------------
// Helpers
// ---------------------------------------------------------------------------
__device__ __forceinline__ float bf2f(unsigned short h) {
    unsigned int u = ((unsigned int)h) << 16;
    return __builtin_bit_cast(float, u);
}
__device__ __forceinline__ unsigned short f2bf(float f) {
    unsigned int u = __builtin_bit_cast(unsigned int, f);
    unsigned int r = u + 0x7FFFu + ((u >> 16) & 1u);   // round-to-nearest-even
    return (unsigned short)(r >> 16);
}
__device__ __forceinline__ float gelu_f(float v) {
    const float c = 0.7978845608028654f;               // sqrt(2/pi)
    float u = c * (v + 0.044715f * v * v * v);
    return 0.5f * v * (1.0f + tanhf(u));
}

// CDNA5 async global->LDS copy (ASYNCcnt-tracked), 16B per lane.
__device__ __forceinline__ void async_copy_b128(unsigned lds_off, const void* gptr) {
    asm volatile("global_load_async_to_lds_b128 %0, %1, off"
                 :: "v"(lds_off), "v"(gptr) : "memory");
}

// ---------------------------------------------------------------------------
// fp32 -> bf16 conversion (plain)
// ---------------------------------------------------------------------------
__global__ __launch_bounds__(256) void cvt_bf16_kernel(const float* __restrict__ in,
                                                       unsigned short* __restrict__ out,
                                                       int n) {
    int i = blockIdx.x * 256 + threadIdx.x;
    if (i < n) out[i] = f2bf(in[i]);
}

// ---------------------------------------------------------------------------
// fp32 -> bf16 with per-layer transpose: out[l][n][k] = in[l][k][n]
// (weights pre-transposed so GEMM B-tiles are k-contiguous per column)
// ---------------------------------------------------------------------------
__global__ __launch_bounds__(256) void cvt_t_kernel(const float* __restrict__ in,
                                                    unsigned short* __restrict__ out,
                                                    int K, int N, int total) {
    int idx = blockIdx.x * 256 + threadIdx.x;
    if (idx >= total) return;
    int kn  = K * N;
    int l   = idx / kn;
    int rem = idx - l * kn;
    int n   = rem / K;
    int k   = rem - n * K;
    out[idx] = f2bf(in[(size_t)l * kn + (size_t)k * N + n]);
}

// ---------------------------------------------------------------------------
// Patchify: img (B,3,224,224) fp32 -> A (B*196, 768) bf16
// ---------------------------------------------------------------------------
__global__ __launch_bounds__(256) void patchify_kernel(const float* __restrict__ img,
                                                       unsigned short* __restrict__ A) {
    int idx = blockIdx.x * 256 + threadIdx.x;
    const int total = PROWS * VIT_E;
    if (idx >= total) return;
    int col = idx % VIT_E;
    int row = idx / VIT_E;
    int n   = row % VIT_N;
    int b   = row / VIT_N;
    int c   = col >> 8;
    int iy  = (col >> 4) & 15;
    int ix  = col & 15;
    int py  = n / 14, px = n % 14;
    size_t src = (((size_t)b * 3 + c) * 224 + (py * 16 + iy)) * 224 + (px * 16 + ix);
    A[idx] = f2bf(img[src]);
}

// ---------------------------------------------------------------------------
// Assemble: x (B,S,E) fp32 = concat(cls, patch_out) + pos_emb
// ---------------------------------------------------------------------------
__global__ __launch_bounds__(256) void assemble_kernel(const float* __restrict__ patch_out,
                                                       const float* __restrict__ cls_tok,
                                                       const float* __restrict__ pos_emb,
                                                       float* __restrict__ x) {
    int idx = blockIdx.x * 256 + threadIdx.x;
    const int total = ROWS * VIT_E;
    if (idx >= total) return;
    int col = idx % VIT_E;
    int row = idx / VIT_E;
    int s   = row % VIT_S;
    int b   = row / VIT_S;
    float v = (s == 0) ? cls_tok[col]
                       : patch_out[((size_t)b * VIT_N + (s - 1)) * VIT_E + col];
    x[idx] = v + pos_emb[(size_t)s * VIT_E + col];
}

// ---------------------------------------------------------------------------
// LayerNorm (no affine): x (rows,768) fp32 -> out (rows,768) bf16
// ---------------------------------------------------------------------------
__global__ __launch_bounds__(256) void ln_kernel(const float* __restrict__ x,
                                                 unsigned short* __restrict__ out,
                                                 int rows) {
    int r = blockIdx.x;
    if (r >= rows) return;
    int t = threadIdx.x;
    const float* xr = x + (size_t)r * VIT_E;
    float a0 = xr[t], a1 = xr[t + 256], a2 = xr[t + 512];
    __shared__ float red[256];
    red[t] = a0 + a1 + a2;
    __syncthreads();
    for (int o = 128; o > 0; o >>= 1) { if (t < o) red[t] += red[t + o]; __syncthreads(); }
    float mu = red[0] * (1.0f / 768.0f);
    __syncthreads();
    float d0 = a0 - mu, d1 = a1 - mu, d2 = a2 - mu;
    red[t] = d0 * d0 + d1 * d1 + d2 * d2;
    __syncthreads();
    for (int o = 128; o > 0; o >>= 1) { if (t < o) red[t] += red[t + o]; __syncthreads(); }
    float inv = rsqrtf(red[0] * (1.0f / 768.0f) + 1e-5f);
    unsigned short* orow = out + (size_t)r * VIT_E;
    orow[t]       = f2bf(d0 * inv);
    orow[t + 256] = f2bf(d1 * inv);
    orow[t + 512] = f2bf(d2 * inv);
}

// ---------------------------------------------------------------------------
// WMMA GEMM: C(MxN) = A(MxK bf16 rm) * Bt(NxK bf16, pre-transposed)
// Tile 64(M) x 256(N) x 32(K); 8 waves in 2x4 grid, each owns 32x64 (8 accs).
// Double-buffered LDS, staged with global_load_async_to_lds_b128 (ASYNCcnt).
// Epilogue compile-time specialized.
// ---------------------------------------------------------------------------
template<bool HAS_BIAS, bool GELU, bool RESID, bool OUTF, bool OUTB>
__global__ __launch_bounds__(256) void gemm_bf16_kernel(
    const unsigned short* __restrict__ A, const unsigned short* __restrict__ Bt,
    const float* __restrict__ bias, const float* __restrict__ resid,
    float* __restrict__ outF, unsigned short* __restrict__ outB,
    int M, int N, int K)
{
    const int lda = K, ldc = N;
    __shared__ __attribute__((aligned(32))) unsigned short As[2][64][40];
    __shared__ __attribute__((aligned(32))) unsigned short Bs[2][256][40];

    int t    = threadIdx.x;
    int wave = t >> 5, lane = t & 31;
    int wm   = wave >> 2;      // 0..1  -> 32-row strip
    int wn   = wave & 3;       // 0..3  -> 64-col strip
    int r    = lane & 15;
    int sel  = lane >> 4;

    int bM = blockIdx.y * 64;
    int bN = blockIdx.x * 256;

    v8f acc[2][4];
    #pragma unroll
    for (int mi = 0; mi < 2; mi++)
        #pragma unroll
        for (int ni = 0; ni < 4; ni++)
            #pragma unroll
            for (int j = 0; j < 8; j++) acc[mi][ni][j] = 0.0f;

    // staging assignments (16B chunks)
    int arow = t >> 2, ach = t & 3;                 // A: 64 rows x 4 chunks
    int aclamp = bM + arow; if (aclamp > M - 1) aclamp = M - 1;   // OOB rows: garbage ok (never stored)
    const unsigned short* aptr = A + (size_t)aclamp * lda + ach * 8;

    auto stage = [&](int kt, int bufi) {
        int k0 = kt * 32;
        async_copy_b128((unsigned)(size_t)&As[bufi][arow][ach * 8], aptr + k0);
        #pragma unroll
        for (int i = 0; i < 4; i++) {
            int idx = t + i * 256;                  // 0..1023
            int col = idx >> 2, ch = idx & 3;       // B: 256 cols x 4 chunks
            const unsigned short* g = Bt + (size_t)(bN + col) * K + k0 + ch * 8;
            async_copy_b128((unsigned)(size_t)&Bs[bufi][col][ch * 8], g);
        }
    };

    const int nk = K >> 5;
    stage(0, 0);

    for (int kt = 0; kt < nk; ++kt) {
        int cur = kt & 1;
        if (kt + 1 < nk) {
            stage(kt + 1, cur ^ 1);
            asm volatile("s_wait_asynccnt 5" ::: "memory");   // tile kt complete (in-order)
        } else {
            asm volatile("s_wait_asynccnt 0" ::: "memory");
        }
        __syncthreads();

        // A fragments (ISA 7.12.2 16-bit layout)
        v16bf afr[2];
        #pragma unroll
        for (int mi = 0; mi < 2; mi++) {
            int row = wm * 32 + mi * 16 + r;
            u16x8 lo = *(const u16x8*)&As[cur][row][sel * 8];        // K 0-7  / 8-15
            u16x8 hi = *(const u16x8*)&As[cur][row][16 + sel * 8];   // K 16-23/ 24-31
            u16x16 cc;
            #pragma unroll
            for (int j = 0; j < 8; j++) { cc[j] = lo[j]; cc[j + 8] = hi[j]; }
            afr[mi] = __builtin_bit_cast(v16bf, cc);
        }
        #pragma unroll
        for (int ni = 0; ni < 4; ni++) {
            int col = wn * 64 + ni * 16 + r;
            u16x8 lo = *(const u16x8*)&Bs[cur][col][sel * 8];         // K 0-7  / 8-15
            u16x8 hi = *(const u16x8*)&Bs[cur][col][16 + sel * 8];    // K 16-23/ 24-31
            u16x16 cc;
            #pragma unroll
            for (int j = 0; j < 8; j++) { cc[j] = lo[j]; cc[j + 8] = hi[j]; }
            v16bf bfr = __builtin_bit_cast(v16bf, cc);
            #pragma unroll
            for (int mi = 0; mi < 2; mi++)
                acc[mi][ni] = __builtin_amdgcn_wmma_f32_16x16x32_bf16(
                    false, afr[mi], false, bfr, (short)0, acc[mi][ni], false, false);
        }
        __syncthreads();
    }

    // Epilogue: C/D layout lane0-15: N=lane, M=vgpr j; lane16-31: M=8+j.
    #pragma unroll
    for (int mi = 0; mi < 2; mi++) {
        #pragma unroll
        for (int ni = 0; ni < 4; ni++) {
            #pragma unroll
            for (int j = 0; j < 8; j++) {
                int lm   = sel * 8 + j;
                int grow = bM + wm * 32 + mi * 16 + lm;
                int gcol = bN + wn * 64 + ni * 16 + r;
                if (grow < M) {
                    float v = acc[mi][ni][j];
                    if constexpr (HAS_BIAS) v += bias[gcol];
                    if constexpr (GELU)     v = gelu_f(v);
                    size_t oidx = (size_t)grow * ldc + gcol;
                    if constexpr (RESID)    v += resid[oidx];
                    if constexpr (OUTF)     outF[oidx] = v;
                    if constexpr (OUTB)     outB[oidx] = f2bf(v);
                }
            }
        }
    }
}

// ---------------------------------------------------------------------------
// Attention: qkv (B*S, 2304) bf16 -> o (B*S, 768) bf16.  One block per (b,h).
// ---------------------------------------------------------------------------
__global__ __launch_bounds__(256) void attn_kernel(const unsigned short* __restrict__ qkv,
                                                   unsigned short* __restrict__ o) {
    const int E3 = 3 * VIT_E;  // 2304
    int bh = blockIdx.x;
    int b  = bh / VIT_H;
    int h  = bh % VIT_H;
    int t  = threadIdx.x, wave = t >> 5, lane = t & 31;

    __shared__ __attribute__((aligned(16))) unsigned short Ks[200][64];
    __shared__ __attribute__((aligned(16))) unsigned short Vs[200][64];
    __shared__ float Ps[8][200];

    size_t base = (size_t)b * VIT_S * E3;
    for (int i = t; i < VIT_S * 64; i += 256) {
        int s = i >> 6, d = i & 63;
        Ks[s][d] = qkv[base + (size_t)s * E3 +  VIT_E    + h * 64 + d];
        Vs[s][d] = qkv[base + (size_t)s * E3 + 2 * VIT_E + h * 64 + d];
    }
    __syncthreads();

    for (int qi = wave; qi < VIT_S; qi += 8) {
        float qf[64];
        const unsigned short* qp = qkv + base + (size_t)qi * E3 + h * 64;
        #pragma unroll
        for (int c = 0; c < 8; c++) {
            u16x8 qq = *(const u16x8*)(qp + c * 8);
            #pragma unroll
            for (int e = 0; e < 8; e++) qf[c * 8 + e] = bf2f(qq[e]);
        }
        float sl[7];
        int cnt = 0;
        float smax = -1e30f;
        for (int j = lane; j < VIT_S; j += 32) {
            float dot = 0.0f;
            #pragma unroll
            for (int c = 0; c < 8; c++) {
                u16x8 kk = *(const u16x8*)&Ks[j][c * 8];
                #pragma unroll
                for (int e = 0; e < 8; e++) dot += qf[c * 8 + e] * bf2f(kk[e]);
            }
            dot *= 0.125f;   // 1/sqrt(64)
            sl[cnt++] = dot;
            smax = fmaxf(smax, dot);
        }
        #pragma unroll
        for (int off = 16; off > 0; off >>= 1) smax = fmaxf(smax, __shfl_xor(smax, off, 32));
        float ssum = 0.0f;
        cnt = 0;
        for (int j = lane; j < VIT_S; j += 32) {
            float p = __expf(sl[cnt++] - smax);
            Ps[wave][j] = p;
            ssum += p;
        }
        #pragma unroll
        for (int off = 16; off > 0; off >>= 1) ssum += __shfl_xor(ssum, off, 32);
        float inv = 1.0f / ssum;
        __threadfence_block();

        int d0 = lane * 2;
        float o0 = 0.0f, o1 = 0.0f;
        for (int j = 0; j < VIT_S; j++) {
            float p = Ps[wave][j];
            unsigned int vv = *(const unsigned int*)&Vs[j][d0];
            o0 += p * bf2f((unsigned short)(vv & 0xffffu));
            o1 += p * bf2f((unsigned short)(vv >> 16));
        }
        unsigned short* op = o + ((size_t)(b * VIT_S + qi)) * VIT_E + h * 64 + d0;
        op[0] = f2bf(o0 * inv);
        op[1] = f2bf(o1 * inv);
    }
}

// ---------------------------------------------------------------------------
// Final: LN(x[:,0])*g+b, then head (768x10). One block per batch.
// ---------------------------------------------------------------------------
__global__ __launch_bounds__(256) void head_kernel(const float* __restrict__ x,
                                                   const float* __restrict__ g,
                                                   const float* __restrict__ bt,
                                                   const float* __restrict__ hw,
                                                   const float* __restrict__ hb,
                                                   float* __restrict__ out) {
    int b = blockIdx.x;
    int t = threadIdx.x;
    __shared__ float row[768];
    __shared__ float red[256];
    const float* xr = x + (size_t)b * VIT_S * VIT_E;   // cls row
    float a0 = xr[t], a1 = xr[t + 256], a2 = xr[t + 512];
    red[t] = a0 + a1 + a2;
    __syncthreads();
    for (int o = 128; o > 0; o >>= 1) { if (t < o) red[t] += red[t + o]; __syncthreads(); }
    float mu = red[0] * (1.0f / 768.0f);
    __syncthreads();
    float d0 = a0 - mu, d1 = a1 - mu, d2 = a2 - mu;
    red[t] = d0 * d0 + d1 * d1 + d2 * d2;
    __syncthreads();
    for (int o = 128; o > 0; o >>= 1) { if (t < o) red[t] += red[t + o]; __syncthreads(); }
    float inv = rsqrtf(red[0] * (1.0f / 768.0f) + 1e-5f);
    __syncthreads();
    row[t]       = d0 * inv * g[t]       + bt[t];
    row[t + 256] = d1 * inv * g[t + 256] + bt[t + 256];
    row[t + 512] = d2 * inv * g[t + 512] + bt[t + 512];
    __syncthreads();
    for (int c = 0; c < 10; c++) {
        float p = row[t]       * hw[(size_t)t * 10 + c]
                + row[t + 256] * hw[(size_t)(t + 256) * 10 + c]
                + row[t + 512] * hw[(size_t)(t + 512) * 10 + c];
        red[t] = p;
        __syncthreads();
        for (int o = 128; o > 0; o >>= 1) { if (t < o) red[t] += red[t + o]; __syncthreads(); }
        if (t == 0) out[b * 10 + c] = red[0] + hb[c];
        __syncthreads();
    }
}

// ---------------------------------------------------------------------------
// Host launch
// ---------------------------------------------------------------------------
extern "C" void kernel_launch(void* const* d_in, const int* in_sizes, int n_in,
                              void* d_out, int out_size, void* d_ws, size_t ws_size,
                              hipStream_t stream) {
    const float* img     = (const float*)d_in[0];
    const float* patch_w = (const float*)d_in[1];
    const float* patch_b = (const float*)d_in[2];
    const float* cls_tok = (const float*)d_in[3];
    const float* pos_emb = (const float*)d_in[4];
    const float* qkv_w   = (const float*)d_in[5];
    const float* fc_w    = (const float*)d_in[6];
    const float* mlp_w1  = (const float*)d_in[7];
    const float* mlp_b1  = (const float*)d_in[8];
    const float* mlp_w2  = (const float*)d_in[9];
    const float* mlp_b2  = (const float*)d_in[10];
    const float* ln_g    = (const float*)d_in[11];
    const float* ln_b    = (const float*)d_in[12];
    const float* head_w  = (const float*)d_in[13];
    const float* head_b  = (const float*)d_in[14];
    float* out = (float*)d_out;

    // workspace carve-up (256B aligned)
    char* ws = (char*)d_ws;
    size_t off = 0;
    auto alloc = [&](size_t bytes) -> char* {
        off = (off + 255) & ~(size_t)255;
        char* p = ws + off;
        off += bytes;
        return p;
    };
    // all weight buffers hold TRANSPOSED bf16 weights: [l][N][K]
    unsigned short* wqT = (unsigned short*)alloc((size_t)VIT_L * 3 * VIT_E * VIT_E * 2);
    unsigned short* wfT = (unsigned short*)alloc((size_t)VIT_L * VIT_E * VIT_E * 2);
    unsigned short* w1T = (unsigned short*)alloc((size_t)VIT_L * VIT_M * VIT_E * 2);
    unsigned short* w2T = (unsigned short*)alloc((size_t)VIT_L * VIT_E * VIT_M * 2);
    unsigned short* wpT = (unsigned short*)alloc((size_t)VIT_E * VIT_E * 2);
    unsigned short* Ab  = (unsigned short*)alloc((size_t)PROWS * VIT_E * 2);
    float*          po  = (float*)         alloc((size_t)PROWS * VIT_E * 4);
    float*          x   = (float*)         alloc((size_t)ROWS * VIT_E * 4);
    unsigned short* hbuf= (unsigned short*)alloc((size_t)ROWS * VIT_E * 2);
    unsigned short* qkvb= (unsigned short*)alloc((size_t)ROWS * 3 * VIT_E * 2);
    unsigned short* ob  = (unsigned short*)alloc((size_t)ROWS * VIT_E * 2);
    unsigned short* tb  = (unsigned short*)alloc((size_t)ROWS * VIT_M * 2);

    auto cvt_t = [&](const float* src, unsigned short* dst, int K, int N, int layers) {
        int total = layers * K * N;
        cvt_t_kernel<<<dim3((unsigned)((total + 255) / 256)), 256, 0, stream>>>(src, dst, K, N, total);
    };

    // 1) weights -> bf16 transposed [l][N][K]
    cvt_t(qkv_w,  wqT, VIT_E, 3 * VIT_E, VIT_L);
    cvt_t(fc_w,   wfT, VIT_E, VIT_E,     VIT_L);
    cvt_t(mlp_w1, w1T, VIT_E, VIT_M,     VIT_L);
    cvt_t(mlp_w2, w2T, VIT_M, VIT_E,     VIT_L);
    cvt_t(patch_w, wpT, VIT_E, VIT_E,    1);

    // 2) patch embed
    {
        size_t tot = (size_t)PROWS * VIT_E;
        patchify_kernel<<<dim3((unsigned)((tot + 255) / 256)), 256, 0, stream>>>(img, Ab);
        // bias + fp32 out
        gemm_bf16_kernel<true, false, false, true, false>
            <<<dim3(VIT_E / 256, (PROWS + 63) / 64), 256, 0, stream>>>(
                Ab, wpT, patch_b, nullptr, po, nullptr, PROWS, VIT_E, VIT_E);
        size_t tot2 = (size_t)ROWS * VIT_E;
        assemble_kernel<<<dim3((unsigned)((tot2 + 255) / 256)), 256, 0, stream>>>(po, cls_tok, pos_emb, x);
    }

    // 3) transformer layers
    for (int l = 0; l < VIT_L; l++) {
        const unsigned short* lwq = wqT + (size_t)l * 3 * VIT_E * VIT_E;
        const unsigned short* lwf = wfT + (size_t)l * VIT_E * VIT_E;
        const unsigned short* lw1 = w1T + (size_t)l * VIT_M * VIT_E;
        const unsigned short* lw2 = w2T + (size_t)l * VIT_E * VIT_M;
        const float* lb1 = mlp_b1 + (size_t)l * VIT_M;
        const float* lb2 = mlp_b2 + (size_t)l * VIT_E;

        ln_kernel<<<dim3(ROWS), 256, 0, stream>>>(x, hbuf, ROWS);
        // qkv: bf16 out only
        gemm_bf16_kernel<false, false, false, false, true>
            <<<dim3(3 * VIT_E / 256, (ROWS + 63) / 64), 256, 0, stream>>>(
                hbuf, lwq, nullptr, nullptr, nullptr, qkvb, ROWS, 3 * VIT_E, VIT_E);
        attn_kernel<<<dim3(VIT_B * VIT_H), 256, 0, stream>>>(qkvb, ob);
        // proj: residual + fp32 out
        gemm_bf16_kernel<false, false, true, true, false>
            <<<dim3(VIT_E / 256, (ROWS + 63) / 64), 256, 0, stream>>>(
                ob, lwf, nullptr, x, x, nullptr, ROWS, VIT_E, VIT_E);
        ln_kernel<<<dim3(ROWS), 256, 0, stream>>>(x, hbuf, ROWS);
        // mlp1: bias + gelu + bf16 out
        gemm_bf16_kernel<true, true, false, false, true>
            <<<dim3(VIT_M / 256, (ROWS + 63) / 64), 256, 0, stream>>>(
                hbuf, lw1, lb1, nullptr, nullptr, tb, ROWS, VIT_M, VIT_E);
        // mlp2: bias + residual + fp32 out
        gemm_bf16_kernel<true, false, true, true, false>
            <<<dim3(VIT_E / 256, (ROWS + 63) / 64), 256, 0, stream>>>(
                tb, lw2, lb2, x, x, nullptr, ROWS, VIT_E, VIT_M);
    }

    // 4) final LN + head
    head_kernel<<<dim3(VIT_B), 256, 0, stream>>>(x, ln_g, ln_b, head_w, head_b, out);
}
